// Decoder_936302871083
// MI455X (gfx1250) — matrix operand order repaired and checked by reading, not compiled, not run on hardware
//
#include <hip/hip_runtime.h>
#include <hip/hip_bf16.h>

typedef __attribute__((ext_vector_type(16))) _Float16 v16h;
typedef __attribute__((ext_vector_type(8)))  _Float16 v8h;
typedef __attribute__((ext_vector_type(4)))  _Float16 v4h;
typedef __attribute__((ext_vector_type(8)))  float    v8f;

#define S_LEN   16384
#define D_MODEL 256
#define NHEAD   8
#define DHEAD   64
#define AHID    512
#define BATCH   2
#define CHUNK   64
#define NCHUNK  (S_LEN / CHUNK)       // 256
#define NTOK    (BATCH * S_LEN)       // 32768
#define NROT    24

__device__ __forceinline__ v8f wmma_f16(v16h a, v16h b, v8f c) {
  return __builtin_amdgcn_wmma_f32_16x16x32_f16(false, a, false, b, (short)0, c, false, false);
}

// A fragment from row-major f16 tile. Requires stride % 8 == 0 (16B-aligned rows).
// Lane pattern (ISA 7.12.2): kb=(lane/16)*8; slots 0-7 -> kb+j, slots 8-15 -> kb+16+j.
__device__ __forceinline__ v16h frag_a_rm(const _Float16* base, int stride, int row,
                                          int kofs, int lane) {
  int kb = kofs + ((lane >> 4) << 3);
  const _Float16* p = base + row * stride + kb;
  v8h lo = *(const v8h*)p;          // ds_load_b128
  v8h hi = *(const v8h*)(p + 16);   // ds_load_b128
  return __builtin_shufflevector(lo, hi, 0,1,2,3,4,5,6,7,8,9,10,11,12,13,14,15);
}
// B fragment from N-major f16 tile Bt[n][k]. Lane: col=n, kb=(lane/16)*8; slots 0-7 -> kb+j, slots 8-15 -> kb+16+j.
__device__ __forceinline__ v16h frag_b_nm(const _Float16* base, int stride, int col,
                                          int kofs, int lane) {
  int kb = kofs + ((lane >> 4) << 3);
  const _Float16* p = base + col * stride + kb;
  v8h lo = *(const v8h*)p;          // ds_load_b128
  v8h hi = *(const v8h*)(p + 16);   // ds_load_b128
  return __builtin_shufflevector(lo, hi, 0,1,2,3,4,5,6,7,8,9,10,11,12,13,14,15);
}

__device__ __forceinline__ float wave_sum(float v) {
#pragma unroll
  for (int off = 16; off; off >>= 1) v += __shfl_xor(v, off, 32);
  return v;
}
__device__ __forceinline__ float wave_max(float v) {
#pragma unroll
  for (int off = 16; off; off >>= 1) v = fmaxf(v, __shfl_xor(v, off, 32));
  return v;
}

// ---------------- embedding: h = x + tile(emb64) + pos ----------------
__global__ __launch_bounds__(256)
void rfmr_embed(const float* __restrict__ x, const float* __restrict__ w1,
                const float* __restrict__ w2, const float* __restrict__ pos1,
                const float* __restrict__ pos2, float* __restrict__ attn,
                float* __restrict__ hidden) {
  size_t gid = (size_t)blockIdx.x * blockDim.x + threadIdx.x;   // NTOK*64 float4 units
  size_t row = gid >> 6;
  int c4 = (int)(gid & 63);
  int d  = c4 * 4;
  int s  = (int)(row % S_LEN);
  int i  = s >> 6, j = s & 63;
  float4 xv = *(const float4*)(x + row * D_MODEL + d);
  float4 e1 = *(const float4*)(w1 + d);
  float4 e2 = *(const float4*)(w2 + (size_t)j * D_MODEL + d);
  float4 pv;
  if (d < 64) pv = *(const float4*)(pos1 + (size_t)i * 64 + d);
  else        pv = *(const float4*)(pos2 + (size_t)j * 192 + (d - 64));
  float4 o;
  o.x = xv.x + e1.x + e2.x + pv.x;
  o.y = xv.y + e1.y + e2.y + pv.y;
  o.z = xv.z + e1.z + e2.z + pv.z;
  o.w = xv.w + e1.w + e2.w + pv.w;
  *(float4*)(attn   + row * D_MODEL + d) = o;
  *(float4*)(hidden + row * D_MODEL + d) = o;
}

// ---------------- layernorm over D=256, one wave per row ----------------
__global__ __launch_bounds__(256)
void rfmr_ln256(const float* __restrict__ x, const float* __restrict__ g,
                const float* __restrict__ bb, float* __restrict__ o) {
  int wave = threadIdx.x >> 5, lane = threadIdx.x & 31;
  size_t row = (size_t)blockIdx.x * 8 + wave;
  const float* xr = x + row * D_MODEL + lane * 8;
  float v[8]; float sum = 0.f;
#pragma unroll
  for (int i = 0; i < 2; ++i) {
    float4 f = *(const float4*)(xr + i * 4);
    v[i*4+0]=f.x; v[i*4+1]=f.y; v[i*4+2]=f.z; v[i*4+3]=f.w;
    sum += f.x + f.y + f.z + f.w;
  }
  float m = wave_sum(sum) * (1.f / 256.f);
  float vs = 0.f;
#pragma unroll
  for (int i = 0; i < 8; ++i) { float d = v[i] - m; vs += d * d; }
  float var = wave_sum(vs) * (1.f / 256.f);
  float rstd = rsqrtf(var + 1e-12f);
  int db = lane * 8;
  float* orow = o + row * D_MODEL + db;
#pragma unroll
  for (int i = 0; i < 8; ++i) orow[i] = (v[i] - m) * rstd * g[db + i] + bb[db + i];
}

// ---------------- final layernorm over concat(attn,hidden) D=512 ----------------
__global__ __launch_bounds__(256)
void rfmr_ln_final(const float* __restrict__ attn, const float* __restrict__ hidden,
                   const float* __restrict__ g, const float* __restrict__ bb,
                   float* __restrict__ out) {
  int wave = threadIdx.x >> 5, lane = threadIdx.x & 31;
  size_t row = (size_t)blockIdx.x * 8 + wave;
  const float* src = (lane < 16) ? (attn + row * 256 + lane * 16)
                                 : (hidden + row * 256 + (lane - 16) * 16);
  float v[16]; float sum = 0.f;
#pragma unroll
  for (int i = 0; i < 4; ++i) {
    float4 f = *(const float4*)(src + i * 4);
    v[i*4+0]=f.x; v[i*4+1]=f.y; v[i*4+2]=f.z; v[i*4+3]=f.w;
    sum += f.x + f.y + f.z + f.w;
  }
  float m = wave_sum(sum) * (1.f / 512.f);
  float vs = 0.f;
#pragma unroll
  for (int i = 0; i < 16; ++i) { float d = v[i] - m; vs += d * d; }
  float rstd = rsqrtf(wave_sum(vs) * (1.f / 512.f) + 1e-12f);
  int db = lane * 16;
  float* orow = out + row * 512 + db;
#pragma unroll
  for (int i = 0; i < 16; ++i) orow[i] = (v[i] - m) * rstd * g[db + i] + bb[db + i];
}

// ---------------- WMMA GEMM: out = [addend +] act(A@W + bias) ----------------
// Block tile 128(M) x 128(N), 8 waves (wm=w>>1 in 0..3, wn=w&1), wave tile 32x64:
// 2 A fragments reused over 4 B fragments -> 8 WMMA per 12 ds_load_b128 per k-step.
// K is compile-time: k-loop fully unrolls (all GEMMs here are K=256 or K=512).
template<int K, bool RELU>
__global__ __launch_bounds__(256)
void rfmr_gemm(const float* __restrict__ A, const float* __restrict__ W,
               const float* __restrict__ bias, const float* __restrict__ addend,
               float* __restrict__ out, int N) {
  __shared__ __align__(16) _Float16 As[128][40];
  __shared__ __align__(16) _Float16 Bt[128][40];
  const int tid = threadIdx.x;
  const int wave = tid >> 5, lane = tid & 31;
  const int wm = wave >> 1, wn = wave & 1;
  const int m_blk = blockIdx.x * 128;
  const int n_blk = blockIdx.y * 128;
  v8f zero = {0.f,0.f,0.f,0.f,0.f,0.f,0.f,0.f};
  v8f acc[2][4] = {{zero, zero, zero, zero}, {zero, zero, zero, zero}};

  const int a_r = tid >> 3, a_c = (tid & 7) * 4;           // A: 4 float4/thread
  const int b_nq = (tid & 31) * 4, b_kq = (tid >> 5) * 4;  // B: 4x4 block/thread

#pragma unroll
  for (int k0 = 0; k0 < K; k0 += 32) {
    if (k0 + 32 < K) {   // WGP-scope prefetch of next tiles (global_prefetch_b8)
      __builtin_prefetch(A + (size_t)(m_blk + a_r) * K + k0 + 32 + a_c, 0, 3);
      __builtin_prefetch(W + (size_t)(k0 + 32 + b_kq) * N + n_blk + b_nq, 0, 3);
    }
    // A tile 128x32 -> As (row-major, v4h stores)
#pragma unroll
    for (int i = 0; i < 4; ++i) {
      int idx = tid + i * 256;
      int r = idx >> 3, c = (idx & 7) * 4;
      float4 f = *(const float4*)(A + (size_t)(m_blk + r) * K + k0 + c);
      v4h h = {(_Float16)f.x, (_Float16)f.y, (_Float16)f.z, (_Float16)f.w};
      *(v4h*)&As[r][c] = h;
    }
    // B tile 32x128 -> Bt (N-major via 4x4 register transpose)
    {
      float rr[4][4];
#pragma unroll
      for (int i = 0; i < 4; ++i) {
        float4 f = *(const float4*)(W + (size_t)(k0 + b_kq + i) * N + n_blk + b_nq);
        rr[i][0] = f.x; rr[i][1] = f.y; rr[i][2] = f.z; rr[i][3] = f.w;
      }
#pragma unroll
      for (int c = 0; c < 4; ++c) {
        v4h h = {(_Float16)rr[0][c], (_Float16)rr[1][c],
                 (_Float16)rr[2][c], (_Float16)rr[3][c]};
        *(v4h*)&Bt[b_nq + c][b_kq] = h;
      }
    }
    __syncthreads();
    v16h af0 = frag_a_rm(&As[0][0], 40, wm * 32 + (lane & 15), 0, lane);
    v16h af1 = frag_a_rm(&As[0][0], 40, wm * 32 + 16 + (lane & 15), 0, lane);
#pragma unroll
    for (int nt = 0; nt < 4; ++nt) {
      v16h bf = frag_b_nm(&Bt[0][0], 40, wn * 64 + nt * 16 + (lane & 15), 0, lane);
      acc[0][nt] = wmma_f16(af0, bf, acc[0][nt]);
      acc[1][nt] = wmma_f16(af1, bf, acc[1][nt]);
    }
    __syncthreads();
  }
#pragma unroll
  for (int mi = 0; mi < 2; ++mi) {
#pragma unroll
    for (int nt = 0; nt < 4; ++nt) {
      int n = n_blk + wn * 64 + nt * 16 + (lane & 15);
#pragma unroll
      for (int r = 0; r < 8; ++r) {
        int m = m_blk + wm * 32 + mi * 16 + r + 8 * (lane >> 4);
        float val = acc[mi][nt][r];
        if (bias)   val += bias[n];
        if (RELU)   val = fmaxf(val, 0.f);
        if (addend) val += addend[(size_t)m * N + n];
        out[(size_t)m * N + n] = val;
      }
    }
  }
}

// ---------------- fused chunked attention (local / LSH) ----------------
// block = (chunk n, head h, batch b), 256 threads, dynamic LDS.
// Ks kept in natural [p][d] layout (= N-major for Q@K^T); Vt transposed to [d][p].
template<bool LSH>
__global__ __launch_bounds__(256)
void rfmr_attn(const float* __restrict__ Q, const float* __restrict__ Kg,
               const float* __restrict__ V, const unsigned int* __restrict__ keys,
               float* __restrict__ O) {
  extern __shared__ char smem[];
  _Float16* Qs = (_Float16*)smem;               // [64][64]        8192 B
  _Float16* Ks = (_Float16*)(smem + 8192);      // [128][72]      18432 B ; reused as P[64][128]
  _Float16* Vt = (_Float16*)(smem + 26624);     // [64][136]      17408 B
  float*    Sc = (float*)(smem + 44032);        // [64][128]      32768 B
  int*      qi = (int*)(smem + 76800);          // [64]
  int*      ki = (int*)(smem + 77056);          // [128]
  float*    rs = (float*)(smem + 77568);        // [128][2]

  const int n = blockIdx.x, h = blockIdx.y, b = blockIdx.z;
  const int pn = (n + NCHUNK - 1) % NCHUNK;
  const int tid = threadIdx.x, wave = tid >> 5, lane = tid & 31;
  const size_t bh = (size_t)b * NHEAD + h;

  // --- stage Q (64x64) row-major ---
  {
    int r = tid >> 2, c0 = (tid & 3) * 16;
    int s = n * CHUNK + r;
    const float* src = LSH ? Q + (bh * S_LEN + s) * DHEAD + c0
                           : Q + ((size_t)b * S_LEN + s) * AHID + h * DHEAD + c0;
#pragma unroll
    for (int i = 0; i < 4; ++i) {
      float4 f = *(const float4*)(src + i * 4);
      v4h hh = {(_Float16)f.x, (_Float16)f.y, (_Float16)f.z, (_Float16)f.w};
      *(v4h*)&Qs[r * 64 + c0 + i * 4] = hh;
    }
  }
  // --- stage V (128x64) transposed -> Vt[d][p], 4p x 8d block per thread ---
  {
    int pb = (tid & 31) * 4, db = (tid >> 5) * 8;
    float rr[4][8];
#pragma unroll
    for (int i = 0; i < 4; ++i) {
      int p = pb + i;
      int s = (p < 64) ? (pn * CHUNK + p) : (n * CHUNK + p - 64);
      const float* src = LSH ? V + (bh * S_LEN + s) * DHEAD + db
                             : V + ((size_t)b * S_LEN + s) * AHID + h * DHEAD + db;
      float4 f0 = *(const float4*)(src);
      float4 f1 = *(const float4*)(src + 4);
      rr[i][0]=f0.x; rr[i][1]=f0.y; rr[i][2]=f0.z; rr[i][3]=f0.w;
      rr[i][4]=f1.x; rr[i][5]=f1.y; rr[i][6]=f1.z; rr[i][7]=f1.w;
    }
#pragma unroll
    for (int j = 0; j < 8; ++j) {
      v4h hh = {(_Float16)rr[0][j], (_Float16)rr[1][j],
                (_Float16)rr[2][j], (_Float16)rr[3][j]};
      *(v4h*)&Vt[(db + j) * 136 + pb] = hh;
    }
  }
  // --- indices ---
  if (tid < 64) {
    int s = n * CHUNK + tid;
    qi[tid] = LSH ? (int)(keys[bh * S_LEN + s] & (S_LEN - 1)) : s;
  }
  if (tid < 128) {
    int s = (tid < 64) ? (pn * CHUNK + tid) : (n * CHUNK + tid - 64);
    ki[tid] = LSH ? (int)(keys[bh * S_LEN + s] & (S_LEN - 1)) : s;
  }
  // --- stage K (128x64) natural row layout with scaling ---
  int kp = tid >> 1, kc0 = (tid & 1) * 32;
  float kv[32];
  {
    int s = (kp < 64) ? (pn * CHUNK + kp) : (n * CHUNK + kp - 64);
    const float* src = LSH ? Kg + (bh * S_LEN + s) * DHEAD + kc0
                           : Kg + ((size_t)b * S_LEN + s) * AHID + h * DHEAD + kc0;
    float ss = 0.f;
#pragma unroll
    for (int i = 0; i < 8; ++i) {
      float4 f = *(const float4*)(src + i * 4);
      kv[i*4+0]=f.x; kv[i*4+1]=f.y; kv[i*4+2]=f.z; kv[i*4+3]=f.w;
      if (LSH) ss += f.x*f.x + f.y*f.y + f.z*f.z + f.w*f.w;
    }
    if (LSH) rs[kp * 2 + (tid & 1)] = ss;
  }
  __syncthreads();
  {
    float scale = 0.125f;                       // DH^-0.5
    if (LSH) {
      float tot = rs[kp * 2] + rs[kp * 2 + 1];
      scale = rsqrtf(tot * (1.f / 64.f) + 1e-6f) * 0.125f;
    }
#pragma unroll
    for (int i = 0; i < 8; ++i) {
      v4h hh = {(_Float16)(kv[i*4+0] * scale), (_Float16)(kv[i*4+1] * scale),
                (_Float16)(kv[i*4+2] * scale), (_Float16)(kv[i*4+3] * scale)};
      *(v4h*)&Ks[kp * 72 + kc0 + i * 4] = hh;
    }
  }
  __syncthreads();

  // --- scores = Q @ K^T : 64x128, 8 waves -> 4 tiles each ---
  {
    int mt = wave >> 1;
    int ntb = (wave & 1) * 4;
    v8f zero = {0.f,0.f,0.f,0.f,0.f,0.f,0.f,0.f};
    v8f sc[4] = {zero, zero, zero, zero};
#pragma unroll
    for (int ks = 0; ks < 64; ks += 32) {
      v16h af = frag_a_rm(Qs, 64, mt * 16 + (lane & 15), ks, lane);
#pragma unroll
      for (int t = 0; t < 4; ++t) {
        v16h bf = frag_b_nm(Ks, 72, (ntb + t) * 16 + (lane & 15), ks, lane);
        sc[t] = wmma_f16(af, bf, sc[t]);
      }
    }
#pragma unroll
    for (int t = 0; t < 4; ++t) {
      int col = (ntb + t) * 16 + (lane & 15);
#pragma unroll
      for (int r = 0; r < 8; ++r)
        Sc[(mt * 16 + r + 8 * (lane >> 4)) * 128 + col] = sc[t][r];
    }
  }
  __syncthreads();

  // --- masked softmax, 8 rows/wave, lane owns 4 contiguous cols; P over Ks ---
  _Float16* Ps = Ks;
#pragma unroll
  for (int rr = 0; rr < 8; ++rr) {
    int r = wave * 8 + rr;
    int myqi = qi[r];
    float4 sv = *(const float4*)&Sc[r * 128 + lane * 4];
    int4  kvi = *(const int4*)&ki[lane * 4];
    float vals[4] = {sv.x, sv.y, sv.z, sv.w};
    int   kk[4]   = {kvi.x, kvi.y, kvi.z, kvi.w};
    float mx = -3.4e38f;
#pragma unroll
    for (int c = 0; c < 4; ++c) {
      float val = (myqi >= kk[c]) ? vals[c] : -1e9f;
      if (LSH && (myqi == kk[c])) val = -1e5f;
      vals[c] = val;
      mx = fmaxf(mx, val);
    }
    mx = wave_max(mx);
    float sum = 0.f;
#pragma unroll
    for (int c = 0; c < 4; ++c) { vals[c] = __expf(vals[c] - mx); sum += vals[c]; }
    sum = wave_sum(sum);
    float inv = 1.f / sum;
    v4h ph = {(_Float16)(vals[0] * inv), (_Float16)(vals[1] * inv),
              (_Float16)(vals[2] * inv), (_Float16)(vals[3] * inv)};
    *(v4h*)&Ps[r * 128 + lane * 4] = ph;
  }
  __syncthreads();

  // --- O = P @ V : 64x64, 8 waves -> 2 tiles each ---
  {
    int mt = wave >> 1;
    int ntb = (wave & 1) * 2;
    v8f zero = {0.f,0.f,0.f,0.f,0.f,0.f,0.f,0.f};
    v8f oc[2] = {zero, zero};
#pragma unroll
    for (int ks = 0; ks < 128; ks += 32) {
      v16h af = frag_a_rm(Ps, 128, mt * 16 + (lane & 15), ks, lane);
#pragma unroll
      for (int t = 0; t < 2; ++t) {
        v16h bf = frag_b_nm(Vt, 136, (ntb + t) * 16 + (lane & 15), ks, lane);
        oc[t] = wmma_f16(af, bf, oc[t]);
      }
    }
#pragma unroll
    for (int t = 0; t < 2; ++t) {
      int d = (ntb + t) * 16 + (lane & 15);
#pragma unroll
      for (int r = 0; r < 8; ++r) {
        int s = n * CHUNK + mt * 16 + r + 8 * (lane >> 4);
        if (LSH) O[(bh * S_LEN + s) * DHEAD + d] = oc[t][r];
        else     O[((size_t)b * S_LEN + s) * AHID + h * DHEAD + d] = oc[t][r];
      }
    }
  }
}

// ---------------- LSH bucket keys: key = bucket*S + s ----------------
__global__ __launch_bounds__(256)
void rfmr_bucket(const float* __restrict__ qk, const float* __restrict__ rot,
                 unsigned int* __restrict__ keys) {
  size_t gid = (size_t)blockIdx.x * blockDim.x + threadIdx.x; // (b*H+h)*S + s
  int s = (int)(gid % S_LEN);
  int h = (int)((gid / S_LEN) % NHEAD);
  int b = (int)(gid / ((size_t)S_LEN * NHEAD));
  const float* q = qk + ((size_t)b * S_LEN + s) * AHID + h * DHEAD;
  float acc[NROT];
#pragma unroll
  for (int j = 0; j < NROT; ++j) acc[j] = 0.f;
  for (int d = 0; d < DHEAD; ++d) {
    float qv = q[d];
    const float* rr = rot + ((size_t)h * DHEAD + d) * NROT;
#pragma unroll
    for (int j = 0; j < NROT; ++j) acc[j] += qv * rr[j];
  }
  int b1 = 0; float best1 = acc[0];
#pragma unroll
  for (int i = 1; i < 16; ++i) {
    float v = (i < 8) ? acc[i] : -acc[i - 8];
    if (v > best1) { best1 = v; b1 = i; }
  }
  int b2 = 0; float best2 = acc[8];
#pragma unroll
  for (int i = 1; i < 32; ++i) {
    float v = (i < 16) ? acc[8 + i] : -acc[8 + i - 16];
    if (v > best2) { best2 = v; b2 = i; }
  }
  unsigned bucket = (unsigned)(b1 + 16 * b2);
  keys[gid] = bucket * (unsigned)S_LEN + (unsigned)s;
}

// ---------------- bitonic sort of 16384 keys per (b,h) in LDS ----------------
__global__ __launch_bounds__(1024)
void rfmr_sort(unsigned int* __restrict__ keys) {
  extern __shared__ unsigned int sk[];
  size_t base = (size_t)blockIdx.x * S_LEN;
  for (int i = threadIdx.x; i < S_LEN; i += 1024) sk[i] = keys[base + i];
  __syncthreads();
  for (int k = 2; k <= S_LEN; k <<= 1) {
    for (int j = k >> 1; j > 0; j >>= 1) {
      for (int i = threadIdx.x; i < S_LEN; i += 1024) {
        int ixj = i ^ j;
        if (ixj > i) {
          unsigned a = sk[i], c = sk[ixj];
          bool up = ((i & k) == 0);
          if ((a > c) == up) { sk[i] = c; sk[ixj] = a; }
        }
      }
      __syncthreads();
    }
  }
  for (int i = threadIdx.x; i < S_LEN; i += 1024) keys[base + i] = sk[i];
}

// ---------------- gather sorted q/v into head-major buffers ----------------
__global__ __launch_bounds__(256)
void rfmr_gather(const float* __restrict__ qk, const float* __restrict__ v,
                 const unsigned int* __restrict__ keys,
                 float* __restrict__ qs, float* __restrict__ vs) {
  size_t gid = (size_t)blockIdx.x * blockDim.x + threadIdx.x; // (bh*S+j)*16 + d4
  int d4 = (int)(gid & 15);
  size_t row = gid >> 4;
  int h = (int)((row / S_LEN) % NHEAD);
  int b = (int)(row / ((size_t)S_LEN * NHEAD));
  int sidx = (int)(keys[row] & (S_LEN - 1));
  size_t src = ((size_t)b * S_LEN + sidx) * AHID + h * DHEAD + d4 * 4;
  size_t dst = row * DHEAD + d4 * 4;
  *(float4*)(qs + dst) = *(const float4*)(qk + src);
  *(float4*)(vs + dst) = *(const float4*)(v + src);
}

// ---------------- scatter sorted attention output back to merged layout ----------------
__global__ __launch_bounds__(256)
void rfmr_scatter(const float* __restrict__ osrt, const unsigned int* __restrict__ keys,
                  float* __restrict__ merged) {
  size_t gid = (size_t)blockIdx.x * blockDim.x + threadIdx.x;
  int d4 = (int)(gid & 15);
  size_t row = gid >> 4;
  int h = (int)((row / S_LEN) % NHEAD);
  int b = (int)(row / ((size_t)S_LEN * NHEAD));
  int sidx = (int)(keys[row] & (S_LEN - 1));
  size_t dst = ((size_t)b * S_LEN + sidx) * AHID + h * DHEAD + d4 * 4;
  *(float4*)(merged + dst) = *(const float4*)(osrt + row * DHEAD + d4 * 4);
}

// ======================= host orchestration =======================
extern "C" void kernel_launch(void* const* d_in, const int* in_sizes, int n_in,
                              void* d_out, int out_size, void* d_ws, size_t ws_size,
                              hipStream_t stream) {
  (void)in_sizes; (void)n_in; (void)out_size; (void)ws_size;
  const float* x        = (const float*)d_in[0];
  const float* fmap_w1  = (const float*)d_in[1];
  const float* fmap_w2  = (const float*)d_in[2];
  const float* pos1     = (const float*)d_in[3];
  const float* pos2     = (const float*)d_in[4];
  const float* ln_out_g = (const float*)d_in[5];
  const float* ln_out_b = (const float*)d_in[6];

  const int layer_is_lsh[6] = {0, 1, 0, 1, 0, 1};
  struct LayerP {
    const float *ln_a_g, *ln_a_b, *Wo, *ln_f_g, *ln_f_b, *W1, *b1, *W2, *b2, *Wv;
    const float *Wq, *Wk, *Wqk;
  } L[6];
  int idx = 7;
  for (int l = 0; l < 6; ++l) {
    L[l].ln_a_g = (const float*)d_in[idx + 0];
    L[l].ln_a_b = (const float*)d_in[idx + 1];
    L[l].Wo     = (const float*)d_in[idx + 2];
    L[l].ln_f_g = (const float*)d_in[idx + 3];
    L[l].ln_f_b = (const float*)d_in[idx + 4];
    L[l].W1     = (const float*)d_in[idx + 5];
    L[l].b1     = (const float*)d_in[idx + 6];
    L[l].W2     = (const float*)d_in[idx + 7];
    L[l].b2     = (const float*)d_in[idx + 8];
    L[l].Wv     = (const float*)d_in[idx + 9];
    if (!layer_is_lsh[l]) {
      L[l].Wq = (const float*)d_in[idx + 10];
      L[l].Wk = (const float*)d_in[idx + 11];
      L[l].Wqk = nullptr;
      idx += 12;
    } else {
      L[l].Wqk = (const float*)d_in[idx + 10];
      L[l].Wq = L[l].Wk = nullptr;
      idx += 11;
    }
  }
  const float* rotations = (const float*)d_in[idx];

  // workspace carve (fp32 elements)
  const size_t N = (size_t)NTOK;
  float* ws = (float*)d_ws;
  float* attn   = ws;  ws += 256 * N;
  float* hidden = ws;  ws += 256 * N;
  float* lnb    = ws;  ws += 256 * N;
  float* tmid   = ws;  ws += 256 * N;
  float* bq     = ws;  ws += 512 * N;
  float* bk     = ws;  ws += 512 * N;
  float* bv     = ws;  ws += 512 * N;
  float* ba     = ws;  ws += 512 * N;
  unsigned int* keys = (unsigned int*)ws;    // B*H*S = 262144 uint32

  const size_t BHS = (size_t)BATCH * NHEAD * S_LEN;
  const unsigned attn_lds = 78592;   // Qs+Ks+Vt+Sc+idx+rowstats

  // embedding -> attn_out = hidden = h
  rfmr_embed<<<dim3((unsigned)((N * 64) / 256)), 256, 0, stream>>>(
      x, fmap_w1, fmap_w2, pos1, pos2, attn, hidden);

  int lsh_i = 0;
  for (int l = 0; l < 6; ++l) {
    const LayerP& P = L[l];
    // a_in = LN(hidden)
    rfmr_ln256<<<dim3((unsigned)(N / 8)), 256, 0, stream>>>(hidden, P.ln_a_g, P.ln_a_b, lnb);

    if (!layer_is_lsh[l]) {
      rfmr_gemm<256,false><<<dim3(256, 4), 256, 0, stream>>>(lnb, P.Wq, nullptr, nullptr, bq, AHID);
      rfmr_gemm<256,false><<<dim3(256, 4), 256, 0, stream>>>(lnb, P.Wk, nullptr, nullptr, bk, AHID);
      rfmr_gemm<256,false><<<dim3(256, 4), 256, 0, stream>>>(lnb, P.Wv, nullptr, nullptr, bv, AHID);
      rfmr_attn<false><<<dim3(NCHUNK, NHEAD, BATCH), 256, attn_lds, stream>>>(bq, bk, bv, nullptr, ba);
      rfmr_gemm<512,false><<<dim3(256, 2), 256, 0, stream>>>(ba, P.Wo, nullptr, attn, attn, D_MODEL);
    } else {
      const float* rot = rotations + (size_t)lsh_i * NHEAD * DHEAD * NROT;
      ++lsh_i;
      rfmr_gemm<256,false><<<dim3(256, 4), 256, 0, stream>>>(lnb, P.Wqk, nullptr, nullptr, bq, AHID);
      rfmr_gemm<256,false><<<dim3(256, 4), 256, 0, stream>>>(lnb, P.Wv, nullptr, nullptr, bv, AHID);
      rfmr_bucket<<<dim3((unsigned)(BHS / 256)), 256, 0, stream>>>(bq, rot, keys);
      rfmr_sort<<<dim3(BATCH * NHEAD), 1024, S_LEN * sizeof(unsigned), stream>>>(keys);
      rfmr_gather<<<dim3((unsigned)(BHS * 16 / 256)), 256, 0, stream>>>(bq, bv, keys, bk, ba);
      // qs in bk (also used as k source), vs in ba; sorted output -> bv
      rfmr_attn<true><<<dim3(NCHUNK, NHEAD, BATCH), 256, attn_lds, stream>>>(bk, bk, ba, keys, bv);
      rfmr_scatter<<<dim3((unsigned)(BHS * 16 / 256)), 256, 0, stream>>>(bv, keys, bq);
      rfmr_gemm<512,false><<<dim3(256, 2), 256, 0, stream>>>(bq, P.Wo, nullptr, attn, attn, D_MODEL);
    }

    // FFN
    rfmr_ln256<<<dim3((unsigned)(N / 8)), 256, 0, stream>>>(attn, P.ln_f_g, P.ln_f_b, lnb);
    rfmr_gemm<256,true ><<<dim3(256, 2), 256, 0, stream>>>(lnb, P.W1, P.b1, nullptr, tmid, 256);
    rfmr_gemm<256,false><<<dim3(256, 2), 256, 0, stream>>>(tmid, P.W2, P.b2, hidden, hidden, 256);
  }

  rfmr_ln_final<<<dim3((unsigned)(N / 8)), 256, 0, stream>>>(attn, hidden, ln_out_g, ln_out_b, (float*)d_out);
}
